// SSDBoxHead_4801773437434
// MI455X (gfx1250) — compile-verified, matrix operand order checked by reading, not run on hardware
//
#include <hip/hip_runtime.h>
#include <hip/hip_bf16.h>
#include <stdint.h>

typedef __attribute__((ext_vector_type(16))) _Float16     v16h;
typedef __attribute__((ext_vector_type(8)))  float        v8f;
typedef __attribute__((ext_vector_type(4)))  unsigned int u32x4;

#define NUM_CLASSES 21
#define CENTER_VAR  0.1f
#define SIZE_VAR    0.2f
#define BATCH       32
#define NUM_LAYERS  6
#define P_TOTAL     8732
#define MT          4        // M-tiles (of 16 rows) per wave
#define ZP_BYTES    4096     // zero page: > max C*2 + 32 bytes (C<=1024)

// Global-address-space load helper: guarantees global_load_b128 (not flat).
__device__ __forceinline__ u32x4 gload4(const void* p) {
#if defined(__HIP_DEVICE_COMPILE__)
    typedef const u32x4 __attribute__((address_space(1)))* g_u32x4_cp;
    return *(g_u32x4_cp)p;
#else
    return *(const u32x4*)p;
#endif
}

// ---------------------------------------------------------------------------
// Kernel 0: zero the 4KB zero page (one block of 256 threads, 16B each)
// ---------------------------------------------------------------------------
__global__ void zero_fill(u32x4* __restrict__ zp) {
    zp[threadIdx.x] = (u32x4){0u, 0u, 0u, 0u};
}

// ---------------------------------------------------------------------------
// Kernel 1: NCHW fp32 -> NHWC f16
// ---------------------------------------------------------------------------
__global__ void nchw_to_nhwc_f16(const float* __restrict__ in,
                                 _Float16* __restrict__ out,
                                 int C, int HW) {
    long total = (long)BATCH * C * HW;
    long i = (long)blockIdx.x * blockDim.x + threadIdx.x;
    if (i >= total) return;
    int p   = (int)(i % HW);
    long t  = i / HW;
    int c   = (int)(t % C);
    int b   = (int)(t / C);
    out[((long)b * HW + p) * C + c] = (_Float16)in[i];
}

// ---------------------------------------------------------------------------
// Kernel 2: pack OIHW fp32 weights into per-lane WMMA B-fragments (f16).
// Layout: [ntile][kchunk][lane(32)][16 f16]; lane's 16 values are
// K = kb+0..7, kb+16..23 with kb = (lane>=16)?8:0, column N = ntile*16 + lane&15.
// ---------------------------------------------------------------------------
__global__ void pack_weights_f16(const float* __restrict__ w,
                                 _Float16* __restrict__ out,
                                 int C, int Nreal, int Npad) {
    int K = 9 * C;
    int numChunks = K >> 5;
    long total = (long)(Npad >> 4) * numChunks * 512;
    long t = (long)blockIdx.x * blockDim.x + threadIdx.x;
    if (t >= total) return;
    int j    = (int)(t & 15);
    int lane = (int)((t >> 4) & 31);
    long blk = t >> 9;
    int kc   = (int)(blk % numChunks);
    int nt   = (int)(blk / numChunks);
    int n    = nt * 16 + (lane & 15);
    int kb   = (lane >= 16) ? 8 : 0;
    int koff = (j < 8) ? (kb + j) : (kb + 16 + (j - 8));
    int k    = (kc << 5) + koff;          // k = tap*C + c (chunks never cross taps)
    int tap  = k / C;
    int c    = k - tap * C;
    float v  = 0.0f;
    if (n < Nreal) v = w[((long)n * C + c) * 9 + tap];
    out[t] = (_Float16)v;
}

// ---------------------------------------------------------------------------
// Kernel 3: implicit-GEMM 3x3 conv via v_wmma_f32_16x16x32_f16.
// One wave = one 64(M)x16(N) tile (4 WMMA M-tiles sharing each B fragment).
// Halo / out-of-range rows read an always-valid zero page -> all loads are
// unconditional global_load_b128 (no exec masking, no zero-fill movs).
// ---------------------------------------------------------------------------
__global__ void conv3x3_wmma_f16(const _Float16* __restrict__ x,
                                 const _Float16* __restrict__ zp,
                                 const _Float16* __restrict__ wp,
                                 const float* __restrict__ bias,
                                 float* __restrict__ out,
                                 int H, int W, int C, int Nreal, int Npad,
                                 int outBase, int perBatch) {
    const int M = BATCH * H * W;
    const int HW = H * W;
    const int tilesM = (M + 16 * MT - 1) / (16 * MT);
    const int tilesN = Npad >> 4;
    const int wave = blockIdx.x * (blockDim.x >> 5) + (threadIdx.x >> 5);
    if (wave >= tilesM * tilesN) return;   // wave-uniform exit (EXEC all-ones below)
    const int lane = threadIdx.x & 31;
    const int tm = wave % tilesM;
    const int tn = wave / tilesM;
    const int numChunks = (9 * C) >> 5;

    const _Float16* __restrict__ wbase =
        wp + ((long)tn * numChunks * 32 + lane) * 16;

    // Per-sub-tile lane rows
    const int kb = (lane >= 16) ? 8 : 0;
    int bArr[MT], yArr[MT], xArr[MT];
    bool rowValid[MT];
#pragma unroll
    for (int i = 0; i < MT; ++i) {
        const int row = tm * (16 * MT) + i * 16 + (lane & 15);
        rowValid[i] = row < M;
        int rr = rowValid[i] ? row : 0;
        bArr[i] = rr / HW;
        int r = rr - bArr[i] * HW;
        yArr[i] = r / W;
        xArr[i] = r - yArr[i] * W;
    }

    v8f acc[MT];
#pragma unroll
    for (int i = 0; i < MT; ++i) acc[i] = (v8f){};

    int kc = 0;
    for (int tap = 0; tap < 9; ++tap) {
        const int dy = tap / 3 - 1;
        const int dx = tap - (tap / 3) * 3 - 1;
        // Per-sub-tile base pointer: real data, or the zero page (always safe)
        const _Float16* ap[MT];
#pragma unroll
        for (int i = 0; i < MT; ++i) {
            const int yy = yArr[i] + dy;
            const int xx = xArr[i] + dx;
            const bool valid =
                rowValid[i] && (yy >= 0) && (yy < H) && (xx >= 0) && (xx < W);
            const int aoff = ((bArr[i] * H + yy) * W + xx) * C + kb;
            ap[i] = valid ? (x + aoff) : zp;   // zp walk stays < ZP_BYTES
        }
        for (int c0 = 0; c0 < C; c0 += 32, ++kc) {
            union { v16h v; u32x4 u[2]; } bfrag;
            const _Float16* bp = wbase + (long)kc * (32 * 16);
            bfrag.u[0] = gload4(bp);
            bfrag.u[1] = gload4(bp + 8);
#pragma unroll
            for (int i = 0; i < MT; ++i) {
                union { v16h v; u32x4 u[2]; } afrag;
                afrag.u[0] = gload4(ap[i] + c0);
                afrag.u[1] = gload4(ap[i] + c0 + 16);
                acc[i] = __builtin_amdgcn_wmma_f32_16x16x32_f16(
                    false, afrag.v, false, bfrag.v, (short)0, acc[i], false, false);
            }
        }
    }

    // D fragment: VGPR r -> row M = r + (lane>=16 ? 8 : 0), col N = lane & 15
    const int n = tn * 16 + (lane & 15);
    if (n < Nreal) {
        const float bv = bias[n];
        const int rbase = (lane >= 16) ? 8 : 0;
#pragma unroll
        for (int i = 0; i < MT; ++i) {
#pragma unroll
            for (int r = 0; r < 8; ++r) {
                const int m = tm * (16 * MT) + i * 16 + rbase + r;
                if (m < M) {
                    const int bb = m / HW;
                    const int pos = m - bb * HW;
                    out[(long)bb * perBatch + outBase + pos * Nreal + n] = acc[i][r] + bv;
                }
            }
        }
    }
}

// ---------------------------------------------------------------------------
// Kernel 4: softmax over 21 classes + bbox decode (center->corner)
// ---------------------------------------------------------------------------
__global__ void head_finalize(const float* __restrict__ logits,
                              const float* __restrict__ reg,
                              const float* __restrict__ priors,
                              float* __restrict__ out) {
    int idx = blockIdx.x * blockDim.x + threadIdx.x;   // over BATCH*P
    if (idx >= BATCH * P_TOTAL) return;
    const float* lg = logits + (long)idx * NUM_CLASSES;
    float mx = lg[0];
#pragma unroll
    for (int i = 1; i < NUM_CLASSES; ++i) mx = fmaxf(mx, lg[i]);
    float e[NUM_CLASSES];
    float s = 0.0f;
#pragma unroll
    for (int i = 0; i < NUM_CLASSES; ++i) { e[i] = __expf(lg[i] - mx); s += e[i]; }
    const float inv = 1.0f / s;
    float* so = out + (long)idx * NUM_CLASSES;
#pragma unroll
    for (int i = 0; i < NUM_CLASSES; ++i) so[i] = e[i] * inv;

    const int p = idx % P_TOTAL;
    const float* pr = priors + (long)p * 4;
    const float* rg = reg + (long)idx * 4;
    const float cx = rg[0] * CENTER_VAR * pr[2] + pr[0];
    const float cy = rg[1] * CENTER_VAR * pr[3] + pr[1];
    const float w  = __expf(rg[2] * SIZE_VAR) * pr[2];
    const float h  = __expf(rg[3] * SIZE_VAR) * pr[3];
    float* bo = out + (long)BATCH * P_TOTAL * NUM_CLASSES + (long)idx * 4;
    bo[0] = cx - w * 0.5f;
    bo[1] = cy - h * 0.5f;
    bo[2] = cx + w * 0.5f;
    bo[3] = cy + h * 0.5f;
}

// ---------------------------------------------------------------------------
// Host launcher
// ---------------------------------------------------------------------------
static const int BPL[NUM_LAYERS] = {4, 6, 6, 6, 4, 4};
static const int OC [NUM_LAYERS] = {512, 1024, 512, 256, 256, 256};
static const int FS [NUM_LAYERS] = {38, 19, 10, 5, 3, 1};

static inline size_t alignUp(size_t v, size_t a) { return (v + a - 1) & ~(a - 1); }

extern "C" void kernel_launch(void* const* d_in, const int* in_sizes, int n_in,
                              void* d_out, int out_size, void* d_ws, size_t ws_size,
                              hipStream_t stream) {
    (void)in_sizes; (void)n_in; (void)out_size; (void)ws_size;
    char* ws = (char*)d_ws;
    size_t cur = 0;

    // workspace layout
    size_t zeroOff = cur; cur = alignUp(cur + ZP_BYTES, 256);
    size_t featOff[NUM_LAYERS], wClsOff[NUM_LAYERS], wRegOff[NUM_LAYERS];
    int nClsPad[NUM_LAYERS], nRegPad[NUM_LAYERS];
    int posOff[NUM_LAYERS];
    int acc = 0;
    for (int l = 0; l < NUM_LAYERS; ++l) { posOff[l] = acc; acc += BPL[l] * FS[l] * FS[l]; }

    for (int l = 0; l < NUM_LAYERS; ++l) {
        size_t n = (size_t)BATCH * OC[l] * FS[l] * FS[l] * sizeof(_Float16);
        featOff[l] = cur; cur = alignUp(cur + n, 256);
    }
    for (int l = 0; l < NUM_LAYERS; ++l) {
        int nc = BPL[l] * NUM_CLASSES;
        int nr = BPL[l] * 4;
        nClsPad[l] = ((nc + 15) / 16) * 16;
        nRegPad[l] = ((nr + 15) / 16) * 16;
        size_t kc = (size_t)nClsPad[l] * 9 * OC[l] * sizeof(_Float16);
        size_t kr = (size_t)nRegPad[l] * 9 * OC[l] * sizeof(_Float16);
        wClsOff[l] = cur; cur = alignUp(cur + kc, 256);
        wRegOff[l] = cur; cur = alignUp(cur + kr, 256);
    }
    size_t logitsOff = cur; cur = alignUp(cur + (size_t)BATCH * P_TOTAL * NUM_CLASSES * sizeof(float), 256);
    size_t regOff    = cur; cur = alignUp(cur + (size_t)BATCH * P_TOTAL * 4 * sizeof(float), 256);

    // 0) zero page
    zero_fill<<<1, ZP_BYTES / 16, 0, stream>>>((u32x4*)(ws + zeroOff));

    // 1) convert features to NHWC f16
    for (int l = 0; l < NUM_LAYERS; ++l) {
        long total = (long)BATCH * OC[l] * FS[l] * FS[l];
        int blocks = (int)((total + 255) / 256);
        nchw_to_nhwc_f16<<<blocks, 256, 0, stream>>>(
            (const float*)d_in[l], (_Float16*)(ws + featOff[l]), OC[l], FS[l] * FS[l]);
    }

    // 2) pack weights (cls + reg)
    for (int l = 0; l < NUM_LAYERS; ++l) {
        {
            int nreal = BPL[l] * NUM_CLASSES;
            long total = (long)(nClsPad[l] >> 4) * ((9 * OC[l]) >> 5) * 512;
            int blocks = (int)((total + 255) / 256);
            pack_weights_f16<<<blocks, 256, 0, stream>>>(
                (const float*)d_in[6 + l], (_Float16*)(ws + wClsOff[l]),
                OC[l], nreal, nClsPad[l]);
        }
        {
            int nreal = BPL[l] * 4;
            long total = (long)(nRegPad[l] >> 4) * ((9 * OC[l]) >> 5) * 512;
            int blocks = (int)((total + 255) / 256);
            pack_weights_f16<<<blocks, 256, 0, stream>>>(
                (const float*)d_in[18 + l], (_Float16*)(ws + wRegOff[l]),
                OC[l], nreal, nRegPad[l]);
        }
    }

    // 3) WMMA implicit-GEMM convs
    for (int l = 0; l < NUM_LAYERS; ++l) {
        const int H = FS[l], W = FS[l], C = OC[l];
        const int M = BATCH * H * W;
        const int tilesM = (M + 16 * MT - 1) / (16 * MT);
        {   // cls head
            int nreal = BPL[l] * NUM_CLASSES;
            int tiles = tilesM * (nClsPad[l] >> 4);
            int blocks = (tiles + 7) / 8;   // 8 waves of 32 per block
            conv3x3_wmma_f16<<<blocks, 256, 0, stream>>>(
                (const _Float16*)(ws + featOff[l]),
                (const _Float16*)(ws + zeroOff),
                (const _Float16*)(ws + wClsOff[l]),
                (const float*)d_in[12 + l],
                (float*)(ws + logitsOff),
                H, W, C, nreal, nClsPad[l],
                posOff[l] * NUM_CLASSES, P_TOTAL * NUM_CLASSES);
        }
        {   // reg head
            int nreal = BPL[l] * 4;
            int tiles = tilesM * (nRegPad[l] >> 4);
            int blocks = (tiles + 7) / 8;
            conv3x3_wmma_f16<<<blocks, 256, 0, stream>>>(
                (const _Float16*)(ws + featOff[l]),
                (const _Float16*)(ws + zeroOff),
                (const _Float16*)(ws + wRegOff[l]),
                (const float*)d_in[24 + l],
                (float*)(ws + regOff),
                H, W, C, nreal, nRegPad[l],
                posOff[l] * 4, P_TOTAL * 4);
        }
    }

    // 4) softmax + box decode -> d_out (scores then boxes, concatenated flat)
    {
        int total = BATCH * P_TOTAL;
        int blocks = (total + 255) / 256;
        head_finalize<<<blocks, 256, 0, stream>>>(
            (const float*)(ws + logitsOff),
            (const float*)(ws + regOff),
            (const float*)d_in[30],
            (float*)d_out);
    }
}
